// LCADecoderLayer_90494960926820
// MI455X (gfx1250) — matrix operand order repaired and checked by reading, not compiled, not run on hardware
//
#include <hip/hip_runtime.h>
#include <math.h>

typedef __bf16 bf16_t;
typedef __bf16 v16bf __attribute__((ext_vector_type(16)));
typedef __bf16 v4bf  __attribute__((ext_vector_type(4)));
typedef float  v8f   __attribute__((ext_vector_type(8)));

// ---- problem constants (match reference) ----
static constexpr int CB    = 2;      // batch
static constexpr int CS    = 2048;   // seq
static constexpr int CD    = 2048;   // hidden
static constexpr int CH    = 16;     // heads
static constexpr int CHD   = 128;    // head dim
static constexpr int CDFF  = 8192;
static constexpr int CDLCA = 4096;
static constexpr int CR    = CB * CS; // 4096 rows

static __device__ __forceinline__ v8f wmma_bf16(v16bf a, v16bf b, v8f c) {
  // D = A(16x32 bf16) * B(32x16 bf16) + C(16x16 f32)
  return __builtin_amdgcn_wmma_f32_16x16x32_bf16(false, a, false, b, (short)0, c,
                                                 false, false);
}

union FragU  { uint4 q[2]; v16bf v; };                 // 32B fragment
union Pack8  { uint4 q;    unsigned short u[8]; };     // 8 halves
union Pack16 { uint4 q[2]; unsigned short u[16]; };    // 16 halves

// ---------------- elementwise kernels (all float4-vectorized) ----------------

__global__ void k_cvt_bf16(const float4* __restrict__ s, v4bf* __restrict__ d, long n4) {
  long i  = (long)blockIdx.x * blockDim.x + threadIdx.x;
  long st = (long)gridDim.x * blockDim.x;
  for (; i < n4; i += st) {
    float4 f = s[i];
    v4bf o; o[0] = (bf16_t)f.x; o[1] = (bf16_t)f.y; o[2] = (bf16_t)f.z; o[3] = (bf16_t)f.w;
    d[i] = o;
  }
}

__global__ void k_fill0(float4* __restrict__ p, long n4) {
  long i  = (long)blockIdx.x * blockDim.x + threadIdx.x;
  long st = (long)gridDim.x * blockDim.x;
  float4 z; z.x = z.y = z.z = z.w = 0.0f;
  for (; i < n4; i += st) p[i] = z;
}

__global__ void k_zero_diag(float* __restrict__ G, int n) {
  int i = blockIdx.x * blockDim.x + threadIdx.x;
  if (i < n) G[(size_t)i * n + i] = 0.0f;
}

__global__ void k_relu_shift(const float4* __restrict__ u, v4bf* __restrict__ a, long n4) {
  long i  = (long)blockIdx.x * blockDim.x + threadIdx.x;
  long st = (long)gridDim.x * blockDim.x;
  for (; i < n4; i += st) {
    float4 f = u[i];
    v4bf o;
    o[0] = (bf16_t)fmaxf(f.x - 0.1f, 0.0f);
    o[1] = (bf16_t)fmaxf(f.y - 0.1f, 0.0f);
    o[2] = (bf16_t)fmaxf(f.z - 0.1f, 0.0f);
    o[3] = (bf16_t)fmaxf(f.w - 0.1f, 0.0f);
    a[i] = o;
  }
}

__global__ void k_lca_update(float4* __restrict__ u, const float4* __restrict__ b,
                             const float4* __restrict__ ag, long n4) {
  // u += DT * ((b - a@G - u) / TAU),  DT/TAU = 0.1
  long i  = (long)blockIdx.x * blockDim.x + threadIdx.x;
  long st = (long)gridDim.x * blockDim.x;
  for (; i < n4; i += st) {
    float4 uv = u[i], bv = b[i], gv = ag[i];
    uv.x += 0.1f * (bv.x - gv.x - uv.x);
    uv.y += 0.1f * (bv.y - gv.y - uv.y);
    uv.z += 0.1f * (bv.z - gv.z - uv.z);
    uv.w += 0.1f * (bv.w - gv.w - uv.w);
    u[i] = uv;
  }
}

__global__ void k_swiglu(const float4* __restrict__ g, const float4* __restrict__ up,
                         v4bf* __restrict__ o, long n4) {
  long i  = (long)blockIdx.x * blockDim.x + threadIdx.x;
  long st = (long)gridDim.x * blockDim.x;
  for (; i < n4; i += st) {
    float4 gv = g[i], uv = up[i];
    v4bf r;
    r[0] = (bf16_t)((gv.x / (1.0f + __expf(-gv.x))) * uv.x);
    r[1] = (bf16_t)((gv.y / (1.0f + __expf(-gv.y))) * uv.y);
    r[2] = (bf16_t)((gv.z / (1.0f + __expf(-gv.z))) * uv.z);
    r[3] = (bf16_t)((gv.w / (1.0f + __expf(-gv.w))) * uv.w);
    o[i] = r;
  }
}

__global__ __launch_bounds__(256)
void k_rmsnorm(const float* __restrict__ x, const float* __restrict__ w,
               bf16_t* __restrict__ y, int d) {
  __shared__ float red[8];
  long row = blockIdx.x;
  const float4* xr = (const float4*)(x + row * (long)d);
  const float4* w4 = (const float4*)w;
  int d4 = d >> 2;
  float s = 0.0f;
  for (int i = threadIdx.x; i < d4; i += 256) {
    float4 v = xr[i];
    s += v.x * v.x + v.y * v.y + v.z * v.z + v.w * v.w;
  }
  for (int o = 16; o > 0; o >>= 1) s += __shfl_down(s, o, 32);
  if ((threadIdx.x & 31) == 0) red[threadIdx.x >> 5] = s;
  __syncthreads();
  if (threadIdx.x == 0) {
    float t = 0.0f;
    for (int i = 0; i < 8; i++) t += red[i];
    red[0] = rsqrtf(t / (float)d + 1e-6f);
  }
  __syncthreads();
  float rs = red[0];
  v4bf* yr = (v4bf*)(y + row * (long)d);
  for (int i = threadIdx.x; i < d4; i += 256) {
    float4 v = xr[i], ww = w4[i];
    v4bf o;
    o[0] = (bf16_t)(v.x * rs * ww.x);
    o[1] = (bf16_t)(v.y * rs * ww.y);
    o[2] = (bf16_t)(v.z * rs * ww.z);
    o[3] = (bf16_t)(v.w * rs * ww.w);
    yr[i] = o;
  }
}

__global__ void k_rope(bf16_t* __restrict__ q, bf16_t* __restrict__ k, long n) {
  long idx = (long)blockIdx.x * blockDim.x + threadIdx.x;
  if (idx >= n) return;
  int  i   = (int)(idx & 63);      // rotation pair index 0..63
  long t   = idx >> 6;
  int  h   = (int)(t & (CH - 1));
  long row = t >> 4;               // 0..CR-1
  int  s   = (int)(row & (CS - 1));
  float inv = powf(10000.0f, -(2.0f * (float)i) / (float)CHD);
  float ang = (float)s * inv;
  float c = cosf(ang), sn = sinf(ang);
  size_t base = (size_t)row * CD + (size_t)h * CHD;
  float a0 = (float)q[base + i], a1 = (float)q[base + i + 64];
  q[base + i]      = (bf16_t)(a0 * c - a1 * sn);
  q[base + i + 64] = (bf16_t)(a1 * c + a0 * sn);
  float b0 = (float)k[base + i], b1 = (float)k[base + i + 64];
  k[base + i]      = (bf16_t)(b0 * c - b1 * sn);
  k[base + i + 64] = (bf16_t)(b1 * c + b0 * sn);
}

// ---------------- WMMA GEMM ----------------
// C[M,N] = A[M,K] * B[K,N] (+ Cadd), bf16 in, f32 accumulate.
// TA: A stored as [K,M]; TB: B stored as [N,K].
// Block tile 256x128x32, 8 waves in 4x2, wave tile 64x64 (16 WMMAs / K-step,
// 16 ds_load_b128 / K-step -> 1.0 loads per WMMA). Double-buffered LDS:
// stage tile s+1 while computing tile s; one barrier per K-step.
// M mult of 256, N mult of 128, K mult of 32 (guaranteed by caller).
template <bool TA, bool TB>
__global__ __launch_bounds__(256)
void k_gemm(const bf16_t* __restrict__ A, const bf16_t* __restrict__ B,
            const float* __restrict__ Cadd, float* __restrict__ Cf,
            bf16_t* __restrict__ Ch, int M, int N, int K, int lda, int ldb) {
  constexpr int BM = 256, BN = 128, BK = 32;
  constexpr int LAS = 40, LBS = 40;  // padded strides (16B aligned, 20-bank skew)
  __shared__ alignas(16) bf16_t As[2][BM * LAS];
  __shared__ alignas(16) bf16_t Bs[2][BN * LBS];  // [n][k]

  int tid  = threadIdx.x;
  int lane = tid & 31, wave = tid >> 5;
  int wm = wave >> 1, wn = wave & 1;  // 4x2 wave grid; wave tile = 64x64
  int hi = lane >> 4, r = lane & 15;
  int m0 = blockIdx.y * BM, n0 = blockIdx.x * BN;

  v8f acc[4][4];
#pragma unroll
  for (int i = 0; i < 4; i++)
#pragma unroll
    for (int j = 0; j < 4; j++)
#pragma unroll
      for (int e = 0; e < 8; e++) acc[i][j][e] = 0.0f;

  auto stageA = [&](int kb, int buf) {
    if (!TA) {
      // 4 threads per row, 16B each (64B contiguous per row)
      int rr = tid >> 2, k0 = (tid & 3) * 8;
#pragma unroll
      for (int pass = 0; pass < 4; pass++) {
        int mm = pass * 64 + rr;
        *(uint4*)(As[buf] + mm * LAS + k0) =
            *(const uint4*)(A + (size_t)(m0 + mm) * lda + kb + k0);
      }
    } else {  // A stored [K,M]; pack k-pairs per m
      int kp = tid >> 4, mb = (tid & 15) * 16;
      Pack16 lo, hi2;
      const bf16_t* g0 = A + (size_t)(kb + 2 * kp) * lda + m0 + mb;
      const bf16_t* g1 = g0 + lda;
      lo.q[0]  = *(const uint4*)g0;  lo.q[1]  = *(const uint4*)(g0 + 8);
      hi2.q[0] = *(const uint4*)g1;  hi2.q[1] = *(const uint4*)(g1 + 8);
#pragma unroll
      for (int e = 0; e < 16; e++)
        *(unsigned int*)(As[buf] + (mb + e) * LAS + 2 * kp) =
            (unsigned int)lo.u[e] | ((unsigned int)hi2.u[e] << 16);
    }
  };
  auto stageB = [&](int kb, int buf) {
    if (TB) {  // B stored [N,K]: direct contiguous copy
      int nn = tid >> 1, k0 = (tid & 1) * 16;
      const uint4* gp = (const uint4*)(B + (size_t)(n0 + nn) * ldb + kb + k0);
      uint4* sp = (uint4*)(Bs[buf] + nn * LBS + k0);
      sp[0] = gp[0]; sp[1] = gp[1];
    } else {  // B stored [K,N]: pack k-pairs per n
      int kp = tid >> 4, nb = (tid & 15) * 8;
      Pack8 lo, hi2;
      lo.q  = *(const uint4*)(B + (size_t)(kb + 2 * kp) * ldb + n0 + nb);
      hi2.q = *(const uint4*)(B + (size_t)(kb + 2 * kp + 1) * ldb + n0 + nb);
#pragma unroll
      for (int e = 0; e < 8; e++)
        *(unsigned int*)(Bs[buf] + (nb + e) * LBS + 2 * kp) =
            (unsigned int)lo.u[e] | ((unsigned int)hi2.u[e] << 16);
    }
  };

  stageA(0, 0);
  stageB(0, 0);
  __syncthreads();

  int nsteps = K / BK;
  for (int s = 0; s < nsteps; s++) {
    int cur = s & 1, nxt = cur ^ 1;
    if (s + 1 < nsteps) {  // prefetch next tile into the other buffer
      stageA((s + 1) * BK, nxt);
      stageB((s + 1) * BK, nxt);
    }
    FragU fa[4], fb[4];
#pragma unroll
    for (int i = 0; i < 4; i++) {
      const bf16_t* ap = As[cur] + (wm * 64 + i * 16 + r) * LAS + hi * 8;
      fa[i].q[0] = *(const uint4*)ap;
      fa[i].q[1] = *(const uint4*)(ap + 16);
    }
#pragma unroll
    for (int j2 = 0; j2 < 4; j2++) {
      const bf16_t* bp = Bs[cur] + (wn * 64 + j2 * 16 + r) * LBS + hi * 8;
      fb[j2].q[0] = *(const uint4*)bp;
      fb[j2].q[1] = *(const uint4*)(bp + 16);
    }
#pragma unroll
    for (int i = 0; i < 4; i++)
#pragma unroll
      for (int j2 = 0; j2 < 4; j2++)
        acc[i][j2] = wmma_bf16(fa[i].v, fb[j2].v, acc[i][j2]);
    __syncthreads();
  }

#pragma unroll
  for (int i = 0; i < 4; i++)
#pragma unroll
    for (int j2 = 0; j2 < 4; j2++)
#pragma unroll
      for (int j = 0; j < 8; j++) {
        int gm = m0 + wm * 64 + i * 16 + j + hi * 8;
        int gn = n0 + wn * 64 + j2 * 16 + r;
        float v = acc[i][j2][j];
        size_t o = (size_t)gm * N + gn;
        if (Cadd) v += Cadd[o];
        if (Cf) Cf[o] = v;
        if (Ch) Ch[o] = (bf16_t)v;
      }
}

// ---------------- Flash-style causal attention ----------------
// Grid: (S/64, H, B), 128 threads (4 waves); wave w owns q-rows [q0+16w, q0+16w+16).
__global__ __launch_bounds__(128)
void k_attn(const bf16_t* __restrict__ Q, const bf16_t* __restrict__ Km,
            const bf16_t* __restrict__ Vm, bf16_t* __restrict__ O, int S, int D) {
  constexpr int HD = 128;
  constexpr int KS = 136;  // Kt stride (16B aligned)
  constexpr int VS = 40;   // transposed V stride [hd][key]
  constexpr int PS = 40;   // P tile stride
  int h = blockIdx.y, b = blockIdx.z;
  int q0 = blockIdx.x * 64;
  int tid = threadIdx.x, lane = tid & 31, wave = tid >> 5;
  int hi = lane >> 4, r = lane & 15;

  __shared__ alignas(16) bf16_t Kt[32 * KS];    // [key][hd]
  __shared__ alignas(16) bf16_t VtT[128 * VS];  // [hd][key]
  __shared__ alignas(16) bf16_t Pt[4][16 * PS]; // per-wave probs [row][key]
  __shared__ float Sf[4][16 * 33];
  __shared__ float mrow[4][16], lrow[4][16], srow[4][16];

  size_t headoff = (size_t)b * S * D + (size_t)h * HD;

  // preload Q fragments (4 k-slices of HD), vectorized from global
  FragU qf[4];
  {
    const bf16_t* qp = Q + headoff + (size_t)(q0 + wave * 16 + r) * D;
#pragma unroll
    for (int kk = 0; kk < 4; kk++) {
      qf[kk].q[0] = *(const uint4*)(qp + kk * 32 + hi * 8);
      qf[kk].q[1] = *(const uint4*)(qp + kk * 32 + 16 + hi * 8);
    }
  }
  v8f o_[8];
#pragma unroll
  for (int t = 0; t < 8; t++)
#pragma unroll
    for (int e = 0; e < 8; e++) o_[t][e] = 0.0f;
  if (lane < 16) { mrow[wave][lane] = -3.0e38f; lrow[wave][lane] = 0.0f; }
  __syncthreads();

  const float scale = 0.08838834764831845f;  // 1/sqrt(128)

  for (int kb = 0; kb < q0 + 64; kb += 32) {
    // K tile: row-major copy, b128 both sides
    {
      int rowk = tid >> 2, c0 = (tid & 3) * 32;
      const uint4* gp = (const uint4*)(Km + headoff + (size_t)(kb + rowk) * D + c0);
      uint4* sp = (uint4*)(Kt + rowk * KS + c0);
#pragma unroll
      for (int e = 0; e < 4; e++) sp[e] = gp[e];
    }
    // V tile: transpose to [hd][key], packing key-pairs
    {
      int kp = tid >> 3, h0 = (tid & 7) * 16;
      const bf16_t* v0 = Vm + headoff + (size_t)(kb + 2 * kp) * D + h0;
      const bf16_t* v1 = v0 + D;
      Pack16 lo, hi2;
      lo.q[0]  = *(const uint4*)v0;       lo.q[1]  = *(const uint4*)(v0 + 8);
      hi2.q[0] = *(const uint4*)v1;       hi2.q[1] = *(const uint4*)(v1 + 8);
#pragma unroll
      for (int e = 0; e < 16; e++)
        *(unsigned int*)(VtT + (h0 + e) * VS + 2 * kp) =
            (unsigned int)lo.u[e] | ((unsigned int)hi2.u[e] << 16);
    }
    __syncthreads();

    // S = Q * K^T (two 16x16 n-tiles over the 32 keys)
#pragma unroll
    for (int nt = 0; nt < 2; nt++) {
      v8f sc;
#pragma unroll
      for (int e = 0; e < 8; e++) sc[e] = 0.0f;
      const bf16_t* krow = Kt + (nt * 16 + r) * KS + hi * 8;
#pragma unroll
      for (int kk = 0; kk < 4; kk++) {
        FragU fb;
        fb.q[0] = *(const uint4*)(krow + kk * 32);
        fb.q[1] = *(const uint4*)(krow + kk * 32 + 16);
        sc = wmma_bf16(qf[kk].v, fb.v, sc);
      }
#pragma unroll
      for (int j = 0; j < 8; j++)
        Sf[wave][(j + hi * 8) * 33 + nt * 16 + r] = sc[j] * scale;
    }
    __syncthreads();

    // streaming softmax per row (lanes 0..15 of each wave)
    if (lane < 16) {
      int qrow = q0 + wave * 16 + lane;
      float mo = mrow[wave][lane];
      float rmax = mo;
      for (int c = 0; c < 32; c++) {
        int key = kb + c;
        float v = (key <= qrow) ? Sf[wave][lane * 33 + c] : -3.0e38f;
        Sf[wave][lane * 33 + c] = v;
        rmax = fmaxf(rmax, v);
      }
      float scl = __expf(mo - rmax);
      float ls  = lrow[wave][lane] * scl;
      for (int c = 0; c < 32; c += 2) {
        float p0 = __expf(Sf[wave][lane * 33 + c] - rmax);
        float p1 = __expf(Sf[wave][lane * 33 + c + 1] - rmax);
        ls += p0 + p1;
        bf16_t b0 = (bf16_t)p0, b1 = (bf16_t)p1;
        *(unsigned int*)(&Pt[wave][lane * PS + c]) =
            (unsigned int)*(unsigned short*)&b0 |
            ((unsigned int)*(unsigned short*)&b1 << 16);
      }
      mrow[wave][lane] = rmax;
      lrow[wave][lane] = ls;
      srow[wave][lane] = scl;
    }
    __syncthreads();

    // rescale O and accumulate P * V
    FragU pf;
    {
      const bf16_t* pp = &Pt[wave][r * PS + hi * 8];
      pf.q[0] = *(const uint4*)pp;
      pf.q[1] = *(const uint4*)(pp + 16);
    }
#pragma unroll
    for (int t = 0; t < 8; t++) {
#pragma unroll
      for (int j = 0; j < 8; j++) o_[t][j] *= srow[wave][j + hi * 8];
      FragU vf;
      const bf16_t* vp = VtT + (t * 16 + r) * VS + hi * 8;
      vf.q[0] = *(const uint4*)vp;
      vf.q[1] = *(const uint4*)(vp + 16);
      o_[t] = wmma_bf16(pf.v, vf.v, o_[t]);
    }
    __syncthreads();
  }

#pragma unroll
  for (int t = 0; t < 8; t++)
#pragma unroll
    for (int j = 0; j < 8; j++) {
      int row16 = j + hi * 8;
      int qrow  = q0 + wave * 16 + row16;
      float inv = 1.0f / lrow[wave][row16];
      O[headoff + (size_t)qrow * D + t * 16 + r] = (bf16_t)(o_[t][j] * inv);
    }
}

// ---------------- host orchestration ----------------

extern "C" void kernel_launch(void* const* d_in, const int* in_sizes, int n_in,
                              void* d_out, int out_size, void* d_ws, size_t ws_size,
                              hipStream_t stream) {
  (void)in_sizes; (void)n_in; (void)out_size; (void)ws_size;
  const float* x      = (const float*)d_in[0];
  const float* w_in   = (const float*)d_in[1];
  const float* w_lca  = (const float*)d_in[2];
  const float* w_post = (const float*)d_in[3];
  const float* Wq = (const float*)d_in[4];
  const float* Wk = (const float*)d_in[5];
  const float* Wv = (const float*)d_in[6];
  const float* Wo = (const float*)d_in[7];
  const float* Wl = (const float*)d_in[8];
  const float* Wg = (const float*)d_in[9];
  const float* Wu = (const float*)d_in[10];
  const float* Wd = (const float*)d_in[11];
  float* out = (float*)d_out;

  const size_t DD = (size_t)CD * CD;
  const size_t DL = (size_t)CD * CDLCA;
  const size_t DF = (size_t)CD * CDFF;
  const size_t RD = (size_t)CR * CD;
  const size_t RL = (size_t)CR * CDLCA;
  const size_t LL = (size_t)CDLCA * CDLCA;
  const size_t RF = (size_t)CR * CDFF;

  char* base = (char*)d_ws;
  size_t off = 0;
  auto take = [&](size_t bytes) -> char* {
    char* q = base + off;
    off += bytes;
    off = (off + 255) & ~(size_t)255;
    return q;
  };

  bf16_t* WqB = (bf16_t*)take(DD * 2);
  bf16_t* WkB = (bf16_t*)take(DD * 2);
  bf16_t* WvB = (bf16_t*)take(DD * 2);
  bf16_t* WoB = (bf16_t*)take(DD * 2);
  bf16_t* WlB = (bf16_t*)take(DL * 2);
  bf16_t* WgB = (bf16_t*)take(DF * 2);
  bf16_t* WuB = (bf16_t*)take(DF * 2);
  bf16_t* WdB = (bf16_t*)take(DF * 2);
  bf16_t* hbf   = (bf16_t*)take(RD * 2);
  bf16_t* qB    = (bf16_t*)take(RD * 2);
  bf16_t* kB    = (bf16_t*)take(RD * 2);
  bf16_t* vB    = (bf16_t*)take(RD * 2);
  bf16_t* attnB = (bf16_t*)take(RD * 2);
  float*  hres  = (float*)take(RD * 4);
  bf16_t* hnB   = (bf16_t*)take(RD * 2);
  float*  bL    = (float*)take(RL * 4);
  float*  Gf    = (float*)take(LL * 4);
  bf16_t* Gb    = (bf16_t*)take(LL * 2);
  float*  u     = (float*)take(RL * 4);
  bf16_t* aB    = (bf16_t*)take(RL * 2);
  float*  ag    = (float*)take(RL * 4);
  float*  hl    = (float*)take(RD * 4);
  // reuse dead LCA region for MLP (exact fits, all 256B-aligned):
  float*  gate = bL;           // RF*4 == bL(64MB)+Gf(64MB)
  float*  up_  = (float*)Gb;   // RF*4 == Gb(32MB)+u(64MB)+aB(32MB)
  bf16_t* guB  = (bf16_t*)ag;  // RF*2 == ag(64MB)
  bf16_t* mB   = hnB;          // hnB dead after bL gemm

  auto cvt = [&](const float* s, bf16_t* d, size_t n) {
    k_cvt_bf16<<<dim3(2048), 256, 0, stream>>>((const float4*)s, (v4bf*)d, (long)(n / 4));
  };
  auto gemmNN = [&](const bf16_t* A, const bf16_t* B, const float* add, float* cf,
                    bf16_t* ch, int M, int N, int K) {
    k_gemm<false, false><<<dim3(N / 128, M / 256), 256, 0, stream>>>(
        A, B, add, cf, ch, M, N, K, K, N);
  };

  // weights -> bf16
  cvt(Wq, WqB, DD); cvt(Wk, WkB, DD); cvt(Wv, WvB, DD); cvt(Wo, WoB, DD);
  cvt(Wl, WlB, DL); cvt(Wg, WgB, DF); cvt(Wu, WuB, DF); cvt(Wd, WdB, DF);

  // --- attention block ---
  k_rmsnorm<<<CR, 256, 0, stream>>>(x, w_in, hbf, CD);
  gemmNN(hbf, WqB, nullptr, nullptr, qB, CR, CD, CD);
  gemmNN(hbf, WkB, nullptr, nullptr, kB, CR, CD, CD);
  gemmNN(hbf, WvB, nullptr, nullptr, vB, CR, CD, CD);
  {
    long n = (long)CR * CH * (CHD / 2);
    k_rope<<<(unsigned)((n + 255) / 256), 256, 0, stream>>>(qB, kB, n);
  }
  k_attn<<<dim3(CS / 64, CH, CB), 128, 0, stream>>>(qB, kB, vB, attnB, CS, CD);
  gemmNN(attnB, WoB, x, hres, nullptr, CR, CD, CD);  // h = residual + attn@Wo

  // --- LCA ---
  k_rmsnorm<<<CR, 256, 0, stream>>>(hres, w_lca, hnB, CD);
  gemmNN(hnB, WlB, nullptr, bL, nullptr, CR, CDLCA, CD);
  // G = Wl^T @ Wl (A transposed), zero diag, then bf16
  k_gemm<true, false><<<dim3(CDLCA / 128, CDLCA / 256), 256, 0, stream>>>(
      WlB, WlB, nullptr, Gf, nullptr, CDLCA, CDLCA, CD, CDLCA, CDLCA);
  k_zero_diag<<<(CDLCA + 255) / 256, 256, 0, stream>>>(Gf, CDLCA);
  cvt(Gf, Gb, LL);
  k_fill0<<<2048, 256, 0, stream>>>((float4*)u, (long)(RL / 4));
  for (int it = 0; it < 10; it++) {
    k_relu_shift<<<2048, 256, 0, stream>>>((const float4*)u, (v4bf*)aB, (long)(RL / 4));
    gemmNN(aB, Gb, nullptr, ag, nullptr, CR, CDLCA, CDLCA);
    k_lca_update<<<2048, 256, 0, stream>>>((float4*)u, (const float4*)bL,
                                           (const float4*)ag, (long)(RL / 4));
  }
  k_relu_shift<<<2048, 256, 0, stream>>>((const float4*)u, (v4bf*)aB, (long)(RL / 4));
  // h = a @ Wl^T (B transposed) -- replaces residual stream
  k_gemm<false, true><<<dim3(CD / 128, CR / 256), 256, 0, stream>>>(
      aB, WlB, nullptr, hl, nullptr, CR, CD, CDLCA, CDLCA, CDLCA);

  // --- SwiGLU MLP ---
  k_rmsnorm<<<CR, 256, 0, stream>>>(hl, w_post, mB, CD);
  gemmNN(mB, WgB, nullptr, gate, nullptr, CR, CDFF, CD);
  gemmNN(mB, WuB, nullptr, up_, nullptr, CR, CDFF, CD);
  k_swiglu<<<4096, 256, 0, stream>>>((const float4*)gate, (const float4*)up_,
                                     (v4bf*)guB, (long)(RF / 4));
  gemmNN(guB, WdB, hl, out, nullptr, CR, CD, CDFF);  // out = hl + mlp
}
